// StageA_GNN_90056874262573
// MI455X (gfx1250) — matrix operand order repaired and checked by reading, not compiled, not run on hardware
//
#include <hip/hip_runtime.h>

// ---------------------------------------------------------------------------
// StageA_GNN on MI455X (gfx1250, wave32, WMMA).
// Memory-bound (~2 GB HBM traffic vs ~30 GFLOP); all GEMMs routed through
// v_wmma_f32_16x16x32_f16 with f32 accumulation. Weights converted+transposed
// to f16 once per launch (~92 KB, L2-resident). One 16-row tile per wave.
// ---------------------------------------------------------------------------

typedef __attribute__((ext_vector_type(16))) _Float16 v16h;
typedef __attribute__((ext_vector_type(8)))  _Float16 v8h;
typedef __attribute__((ext_vector_type(8)))  float    v8f;

#define DEVINL static __device__ __forceinline__

DEVINL v8f zero8() {
  v8f z;
#pragma unroll
  for (int i = 0; i < 8; ++i) z[i] = 0.0f;
  return z;
}

DEVINL v8f wmma_f16(v16h a, v16h b, v8f c) {
  // D = A(16x32 f16) * B(32x16 f16) + C(16x16 f32)
  return __builtin_amdgcn_wmma_f32_16x16x32_f16(false, a, false, b, (short)0, c,
                                                false, false);
}

// ---- A fragment (16x32 f16, ISA layout) -----------------------------------
// Lane holds row M = lane&15.  k0 = (lane<16 ? 0 : 8).
// Halves 0..7  = K(k0..k0+7), halves 8..15 = K(k0+16..k0+23).
// `p` points at (row_base + kk*32 + k0); must be 16B aligned.
DEVINL v16h fragA_f32(const float* p) {
  const float4* q = (const float4*)p;
  float4 u0 = q[0], u1 = q[1], u2 = q[4], u3 = q[5];
  v16h f;
  f[0]=(_Float16)u0.x;  f[1]=(_Float16)u0.y;  f[2]=(_Float16)u0.z;  f[3]=(_Float16)u0.w;
  f[4]=(_Float16)u1.x;  f[5]=(_Float16)u1.y;  f[6]=(_Float16)u1.z;  f[7]=(_Float16)u1.w;
  f[8]=(_Float16)u2.x;  f[9]=(_Float16)u2.y;  f[10]=(_Float16)u2.z; f[11]=(_Float16)u2.w;
  f[12]=(_Float16)u3.x; f[13]=(_Float16)u3.y; f[14]=(_Float16)u3.z; f[15]=(_Float16)u3.w;
  return f;
}

DEVINL v16h fragA_f32_scaled(const float* p, float s) {
  const float4* q = (const float4*)p;
  float4 u0 = q[0], u1 = q[1], u2 = q[4], u3 = q[5];
  v16h f;
  f[0]=(_Float16)(u0.x*s);  f[1]=(_Float16)(u0.y*s);  f[2]=(_Float16)(u0.z*s);  f[3]=(_Float16)(u0.w*s);
  f[4]=(_Float16)(u1.x*s);  f[5]=(_Float16)(u1.y*s);  f[6]=(_Float16)(u1.z*s);  f[7]=(_Float16)(u1.w*s);
  f[8]=(_Float16)(u2.x*s);  f[9]=(_Float16)(u2.y*s);  f[10]=(_Float16)(u2.z*s); f[11]=(_Float16)(u2.w*s);
  f[12]=(_Float16)(u3.x*s); f[13]=(_Float16)(u3.y*s); f[14]=(_Float16)(u3.z*s); f[15]=(_Float16)(u3.w*s);
  return f;
}

DEVINL v16h fragA_f16(const _Float16* p) {  // same layout, f16 source (LDS tiles)
  v8h lo = *(const v8h*)p;
  v8h hi = *(const v8h*)(p + 16);
  v16h f;
#pragma unroll
  for (int i = 0; i < 8; ++i) { f[i] = lo[i]; f[i + 8] = hi[i]; }
  return f;
}

// ---- B fragment (32x16 f16, ISA layout) -----------------------------------
// Lane holds column N = lane&15; k0 = (lane<16 ? 0 : 8): halves 0..7 = K(k0..k0+7),
// halves 8..15 = K(k0+16..k0+23).  With W pre-transposed to [N][K]:
// p = Wt + n*K + kk*32 + (lane>=16 ? 8 : 0).
DEVINL v16h fragB_f16(const _Float16* p) {
  v8h lo = *(const v8h*)p;
  v8h hi = *(const v8h*)(p + 16);
  v16h f;
#pragma unroll
  for (int i = 0; i < 8; ++i) { f[i] = lo[i]; f[i + 8] = hi[i]; }
  return f;
}

// ---------------------------------------------------------------------------
// Small utility kernels
// ---------------------------------------------------------------------------
__global__ void zero_f32(float* __restrict__ p, int n) {
  int i = blockIdx.x * blockDim.x + threadIdx.x;
  if (i < n) p[i] = 0.0f;
}

__global__ void sigmoid_f32(const float* __restrict__ in, float* __restrict__ out, int n) {
  int i = blockIdx.x * blockDim.x + threadIdx.x;
  if (i < n) out[i] = 1.0f / (1.0f + __expf(-in[i]));
}

// W[K][Nn] f32 row-major -> Wt[Nn][K] f16 (so B fragments are contiguous).
__global__ void cvt_transpose_f16(const float* __restrict__ W, _Float16* __restrict__ Wt,
                                  int K, int Nn) {
  int i = blockIdx.x * blockDim.x + threadIdx.x;
  if (i < K * Nn) {
    int k = i / Nn, n = i - k * Nn;
    Wt[(size_t)n * K + k] = (_Float16)W[i];
  }
}

// ---------------------------------------------------------------------------
// Encoder: h = relu(relu(x @ W1 + b1) @ W2 + b2).  One 16-node tile / wave.
// ---------------------------------------------------------------------------
__global__ __launch_bounds__(256) void encoder_k(
    const float* __restrict__ x, const _Float16* __restrict__ w1t,
    const float* __restrict__ b1, const _Float16* __restrict__ w2t,
    const float* __restrict__ b2, float* __restrict__ h, int nTiles) {
  __shared__ __align__(16) _Float16 t1[8][16 * 80];  // padded stride 80 halves
  const int lane = threadIdx.x & 31;
  const int wid  = threadIdx.x >> 5;
  const int r = lane & 15, halfw = lane >> 4, k0 = halfw ? 8 : 0;
  const int t = blockIdx.x * 8 + wid;
  if (t >= nTiles) return;

  const float* xrow = x + (size_t)(t * 16 + r) * 256;
  v16h a[8];
#pragma unroll
  for (int kk = 0; kk < 8; ++kk) a[kk] = fragA_f32(xrow + kk * 32 + k0);

  _Float16* tile = &t1[wid][0];
#pragma unroll
  for (int nt = 0; nt < 4; ++nt) {
    const int n = nt * 16 + r;
    const _Float16* wp = w1t + (size_t)n * 256 + halfw * 8;
    v8f acc = zero8();
#pragma unroll
    for (int kk = 0; kk < 8; ++kk) acc = wmma_f16(a[kk], fragB_f16(wp + kk * 32), acc);
    const float bias = b1[n];
#pragma unroll
    for (int v = 0; v < 8; ++v) {
      float z = acc[v] + bias;
      z = z > 0.0f ? z : 0.0f;
      tile[(v + 8 * halfw) * 80 + n] = (_Float16)z;   // row=M, col=N
    }
  }
  __builtin_amdgcn_wave_barrier();  // LDS in-order within wave; block reordering

  v16h c0 = fragA_f16(tile + r * 80 + 0 + k0);
  v16h c1 = fragA_f16(tile + r * 80 + 32 + k0);
#pragma unroll
  for (int nt = 0; nt < 4; ++nt) {
    const int n = nt * 16 + r;
    const _Float16* wp = w2t + (size_t)n * 64 + halfw * 8;
    v8f acc = zero8();
    acc = wmma_f16(c0, fragB_f16(wp + 0),  acc);
    acc = wmma_f16(c1, fragB_f16(wp + 32), acc);
    const float bias = b2[n];
#pragma unroll
    for (int v = 0; v < 8; ++v) {
      float z = acc[v] + bias;
      z = z > 0.0f ? z : 0.0f;
      h[(size_t)(t * 16 + v + 8 * halfw) * 64 + n] = z;
    }
  }
}

// ---------------------------------------------------------------------------
// Edge gate + weighted scatter.  One 16-edge tile per wave.
// A = [h_src | h_dst] (16x128 f16), B = gate_w1^T; 16 WMMA per tile.
// ---------------------------------------------------------------------------
__global__ __launch_bounds__(256) void edge_k(
    const float* __restrict__ h, const int* __restrict__ src,
    const int* __restrict__ dst, const float* __restrict__ bw,
    const float* __restrict__ rho, const _Float16* __restrict__ gw1t,
    const float* __restrict__ gb1, const float* __restrict__ gw2,
    const float* __restrict__ gb2, float* __restrict__ mbuf,
    float* __restrict__ deg, int nTiles) {
  __shared__ float sdot[8][16];
  const int lane = threadIdx.x & 31;
  const int wid  = threadIdx.x >> 5;
  const int r = lane & 15, halfw = lane >> 4, k0 = halfw ? 8 : 0;
  const int t = blockIdx.x * 8 + wid;
  if (t >= nTiles) return;

  const int e  = t * 16 + r;
  const int is = src[e], id = dst[e];
  const float* hs = h + (size_t)is * 64;
  const float* hd = h + (size_t)id * 64;

  v16h a0 = fragA_f32(hs + 0  + k0);
  v16h a1 = fragA_f32(hs + 32 + k0);
  v16h a2 = fragA_f32(hd + 0  + k0);
  v16h a3 = fragA_f32(hd + 32 + k0);

  float pv[8];
#pragma unroll
  for (int v = 0; v < 8; ++v) pv[v] = 0.0f;

#pragma unroll
  for (int nt = 0; nt < 4; ++nt) {
    const int n = nt * 16 + r;
    const _Float16* wp = gw1t + (size_t)n * 128 + halfw * 8;
    v8f acc = zero8();
    acc = wmma_f16(a0, fragB_f16(wp + 0),  acc);
    acc = wmma_f16(a1, fragB_f16(wp + 32), acc);
    acc = wmma_f16(a2, fragB_f16(wp + 64), acc);
    acc = wmma_f16(a3, fragB_f16(wp + 96), acc);
    const float bias = gb1[n], w2n = gw2[n];
#pragma unroll
    for (int v = 0; v < 8; ++v) {
      float z = acc[v] + bias;
      z = z > 0.0f ? z : 0.0f;
      pv[v] += z * w2n;     // partial dot(z, gate_w2) over this lane's column
    }
  }
  // reduce over the 16 N-columns (xor butterfly stays within each half-wave)
#pragma unroll
  for (int mk = 1; mk < 16; mk <<= 1)
#pragma unroll
    for (int v = 0; v < 8; ++v) pv[v] += __shfl_xor(pv[v], mk, 32);

  if (r == 0) {
#pragma unroll
    for (int v = 0; v < 8; ++v) sdot[wid][halfw * 8 + v] = pv[v];  // edge m=v+8*halfw
  }
  __builtin_amdgcn_wave_barrier();  // wave-private LDS, DS ops are in-order
  const float dotr = sdot[wid][r];

  const float gate = 1.0f / (1.0f + __expf(-(dotr + gb2[0])));
  const float wgt  = bw[e] * gate * rho[is] * rho[id];

  float* mr = mbuf + (size_t)id * 64 + halfw * 32;
  const float* hp = hs + halfw * 32;  // message uses h_src (L0-resident from A load)
#pragma unroll
  for (int j = 0; j < 32; ++j) atomicAdd(mr + j, wgt * hp[j]);
  if (halfw == 0) atomicAdd(deg + id, wgt);
}

// ---------------------------------------------------------------------------
// Node update + LayerNorm.  neigh = m/(deg+eps); u = relu(neigh@W1+b1)@W2+b2;
// h = LN(h + u).  One 16-node tile per wave.
// ---------------------------------------------------------------------------
__global__ __launch_bounds__(256) void update_k(
    const float* __restrict__ mbuf, const float* __restrict__ deg,
    float* __restrict__ h, const _Float16* __restrict__ w1t,
    const float* __restrict__ b1, const _Float16* __restrict__ w2t,
    const float* __restrict__ b2, const float* __restrict__ lng,
    const float* __restrict__ lnb, int nTiles) {
  __shared__ __align__(16) _Float16 tt[8][16 * 80];
  const int lane = threadIdx.x & 31;
  const int wid  = threadIdx.x >> 5;
  const int r = lane & 15, halfw = lane >> 4, k0 = halfw ? 8 : 0;
  const int t = blockIdx.x * 8 + wid;
  if (t >= nTiles) return;

  const int row = t * 16 + r;
  const float invd = 1.0f / (deg[row] + 1e-8f);
  const float* mrow = mbuf + (size_t)row * 64;
  v16h a0 = fragA_f32_scaled(mrow + 0  + k0, invd);
  v16h a1 = fragA_f32_scaled(mrow + 32 + k0, invd);

  _Float16* tile = &tt[wid][0];
#pragma unroll
  for (int nt = 0; nt < 4; ++nt) {
    const int n = nt * 16 + r;
    const _Float16* wp = w1t + (size_t)n * 64 + halfw * 8;
    v8f acc = zero8();
    acc = wmma_f16(a0, fragB_f16(wp + 0),  acc);
    acc = wmma_f16(a1, fragB_f16(wp + 32), acc);
    const float bias = b1[n];
#pragma unroll
    for (int v = 0; v < 8; ++v) {
      float z = acc[v] + bias;
      z = z > 0.0f ? z : 0.0f;
      tile[(v + 8 * halfw) * 80 + n] = (_Float16)z;
    }
  }
  __builtin_amdgcn_wave_barrier();

  v16h c0 = fragA_f16(tile + r * 80 + 0 + k0);
  v16h c1 = fragA_f16(tile + r * 80 + 32 + k0);

  float pre[4][8];
  float sum[8], ssq[8];
#pragma unroll
  for (int v = 0; v < 8; ++v) { sum[v] = 0.0f; ssq[v] = 0.0f; }

#pragma unroll
  for (int nt = 0; nt < 4; ++nt) {
    const int n = nt * 16 + r;
    const _Float16* wp = w2t + (size_t)n * 64 + halfw * 8;
    v8f acc = zero8();
    acc = wmma_f16(c0, fragB_f16(wp + 0),  acc);
    acc = wmma_f16(c1, fragB_f16(wp + 32), acc);
    const float bias = b2[n];
#pragma unroll
    for (int v = 0; v < 8; ++v) {
      const float p = h[(size_t)(t * 16 + v + 8 * halfw) * 64 + n] + acc[v] + bias;
      pre[nt][v] = p;
      sum[v] += p;
      ssq[v] += p * p;
    }
  }
  // row statistics: reduce across the 16 N-lanes of each half-wave
#pragma unroll
  for (int mk = 1; mk < 16; mk <<= 1)
#pragma unroll
    for (int v = 0; v < 8; ++v) {
      sum[v] += __shfl_xor(sum[v], mk, 32);
      ssq[v] += __shfl_xor(ssq[v], mk, 32);
    }

#pragma unroll
  for (int nt = 0; nt < 4; ++nt) {
    const int n = nt * 16 + r;
    const float g = lng[n], bb = lnb[n];
#pragma unroll
    for (int v = 0; v < 8; ++v) {
      const float mean = sum[v] * (1.0f / 64.0f);
      const float var  = ssq[v] * (1.0f / 64.0f) - mean * mean;
      const float rs   = rsqrtf(var + 1e-5f);
      h[(size_t)(t * 16 + v + 8 * halfw) * 64 + n] = (pre[nt][v] - mean) * rs * g + bb;
    }
  }
}

// ---------------------------------------------------------------------------
// toU head: U = softplus(h @ toU_w + toU_b), [N,32].
// ---------------------------------------------------------------------------
__global__ __launch_bounds__(256) void tou_k(
    const float* __restrict__ h, const _Float16* __restrict__ wt,
    const float* __restrict__ bias, float* __restrict__ U, int nTiles) {
  const int lane = threadIdx.x & 31;
  const int wid  = threadIdx.x >> 5;
  const int r = lane & 15, halfw = lane >> 4, k0 = halfw ? 8 : 0;
  const int t = blockIdx.x * 8 + wid;
  if (t >= nTiles) return;

  const float* hrow = h + (size_t)(t * 16 + r) * 64;
  v16h a0 = fragA_f32(hrow + 0  + k0);
  v16h a1 = fragA_f32(hrow + 32 + k0);

#pragma unroll
  for (int nt = 0; nt < 2; ++nt) {
    const int n = nt * 16 + r;
    const _Float16* wp = wt + (size_t)n * 64 + halfw * 8;
    v8f acc = zero8();
    acc = wmma_f16(a0, fragB_f16(wp + 0),  acc);
    acc = wmma_f16(a1, fragB_f16(wp + 32), acc);
    const float b = bias[n];
#pragma unroll
    for (int v = 0; v < 8; ++v) {
      const float s  = acc[v] + b;
      const float sp = s > 20.0f ? s : log1pf(__expf(s));
      U[(size_t)(t * 16 + v + 8 * halfw) * 32 + n] = sp;
    }
  }
}

// ---------------------------------------------------------------------------
// Launcher
// ---------------------------------------------------------------------------
extern "C" void kernel_launch(void* const* d_in, const int* in_sizes, int n_in,
                              void* d_out, int out_size, void* d_ws, size_t ws_size,
                              hipStream_t stream) {
  (void)in_sizes; (void)n_in; (void)out_size; (void)ws_size;
  constexpr int N = 50000, C = 256, E = 800000, H = 64, K = 32, GH = 64, L = 2;

  const float* x       = (const float*)d_in[0];
  const int*   src     = (const int*)  d_in[1];
  const int*   dst     = (const int*)  d_in[2];
  const float* base_w  = (const float*)d_in[3];
  const float* enc_w1  = (const float*)d_in[4];
  const float* enc_b1  = (const float*)d_in[5];
  const float* enc_w2  = (const float*)d_in[6];
  const float* enc_b2  = (const float*)d_in[7];
  const float* gate_w1 = (const float*)d_in[8];
  const float* gate_b1 = (const float*)d_in[9];
  const float* gate_w2 = (const float*)d_in[10];
  const float* gate_b2 = (const float*)d_in[11];
  const float* upd_w1  = (const float*)d_in[12];
  const float* upd_b1  = (const float*)d_in[13];
  const float* upd_w2  = (const float*)d_in[14];
  const float* upd_b2  = (const float*)d_in[15];
  const float* ln_g    = (const float*)d_in[16];
  const float* ln_b    = (const float*)d_in[17];
  const float* rho_raw = (const float*)d_in[18];
  const float* toU_w   = (const float*)d_in[19];
  const float* toU_b   = (const float*)d_in[20];
  float* U = (float*)d_out;

  char* ws = (char*)d_ws;
  size_t off = 0;
  auto alloc = [&](size_t bytes) -> void* {
    off = (off + 255) & ~(size_t)255;
    void* p = ws + off;
    off += bytes;
    return p;
  };
  float*     h        = (float*)    alloc((size_t)N * H * 4);
  float*     mbuf     = (float*)    alloc((size_t)N * H * 4);
  float*     deg      = (float*)    alloc((size_t)N * 4);
  float*     rho      = (float*)    alloc((size_t)N * 4);
  _Float16*  enc_w1t  = (_Float16*) alloc((size_t)C * H * 2);
  _Float16*  enc_w2t  = (_Float16*) alloc((size_t)H * H * 2);
  _Float16*  gate_w1t = (_Float16*) alloc((size_t)2 * H * GH * 2);
  _Float16*  upd_w1t  = (_Float16*) alloc((size_t)L * H * H * 2);
  _Float16*  upd_w2t  = (_Float16*) alloc((size_t)L * H * H * 2);
  _Float16*  toU_wt   = (_Float16*) alloc((size_t)H * K * 2);

  const int TP = 256;
  auto cdiv = [](int a, int b) { return (a + b - 1) / b; };

  // f32 -> f16 transposed weight copies (tiny, one-time per launch)
  cvt_transpose_f16<<<cdiv(C * H, TP), TP, 0, stream>>>(enc_w1, enc_w1t, C, H);
  cvt_transpose_f16<<<cdiv(H * H, TP), TP, 0, stream>>>(enc_w2, enc_w2t, H, H);
  cvt_transpose_f16<<<cdiv(2 * H * GH, TP), TP, 0, stream>>>(gate_w1, gate_w1t, 2 * H, GH);
  for (int l = 0; l < L; ++l) {
    cvt_transpose_f16<<<cdiv(H * H, TP), TP, 0, stream>>>(upd_w1 + (size_t)l * H * H,
                                                          upd_w1t + (size_t)l * H * H, H, H);
    cvt_transpose_f16<<<cdiv(H * H, TP), TP, 0, stream>>>(upd_w2 + (size_t)l * H * H,
                                                          upd_w2t + (size_t)l * H * H, H, H);
  }
  cvt_transpose_f16<<<cdiv(H * K, TP), TP, 0, stream>>>(toU_w, toU_wt, H, K);

  sigmoid_f32<<<cdiv(N, TP), TP, 0, stream>>>(rho_raw, rho, N);

  const int nodeTiles = N / 16;                  // 3125 (exact)
  const int edgeTiles = E / 16;                  // 50000 (exact)
  const int nodeGrid  = cdiv(nodeTiles, 8);      // 8 waves/block
  const int edgeGrid  = cdiv(edgeTiles, 8);

  encoder_k<<<nodeGrid, TP, 0, stream>>>(x, enc_w1t, enc_b1, enc_w2t, enc_b2, h, nodeTiles);

  for (int l = 0; l < L; ++l) {
    zero_f32<<<cdiv(N * H, TP), TP, 0, stream>>>(mbuf, N * H);
    zero_f32<<<cdiv(N, TP), TP, 0, stream>>>(deg, N);
    edge_k<<<edgeGrid, TP, 0, stream>>>(h, src, dst, base_w, rho, gate_w1t, gate_b1,
                                        gate_w2, gate_b2, mbuf, deg, edgeTiles);
    update_k<<<nodeGrid, TP, 0, stream>>>(mbuf, deg, h,
                                          upd_w1t + (size_t)l * H * H, upd_b1 + (size_t)l * H,
                                          upd_w2t + (size_t)l * H * H, upd_b2 + (size_t)l * H,
                                          ln_g + (size_t)l * H, ln_b + (size_t)l * H, nodeTiles);
  }

  tou_k<<<nodeGrid, TP, 0, stream>>>(h, toU_wt, toU_b, U, nodeTiles);
}
